// VQProsodyEncoder_81896436400205
// MI455X (gfx1250) — matrix operand order repaired and checked by reading, not compiled, not run on hardware
//
#include <hip/hip_runtime.h>

typedef __attribute__((ext_vector_type(16))) __bf16          v16bf;
typedef __attribute__((ext_vector_type(16))) unsigned short  v16u;
typedef __attribute__((ext_vector_type(8)))  float           v8f;
typedef __attribute__((ext_vector_type(2)))  float           v2f;

static __device__ __forceinline__ unsigned short f2bf(float f) {
    unsigned int u = __builtin_bit_cast(unsigned int, f);
    u += 0x7fffu + ((u >> 16) & 1u);           // round to nearest even
    return (unsigned short)(u >> 16);
}
static __device__ __forceinline__ float bf2f(unsigned short h) {
    unsigned int u = ((unsigned int)h) << 16;
    return __builtin_bit_cast(float, u);
}

// gfx1250 async global->LDS copy (ASYNCcnt-tracked), 16 bytes per lane.
static __device__ __forceinline__ void async_load_b128(unsigned lds_off,
                                                       unsigned long long gaddr) {
    asm volatile("global_load_async_to_lds_b128 %0, %1, off"
                 :: "v"(lds_off), "v"(gaddr) : "memory");
}
static __device__ __forceinline__ void wait_asynccnt0() {
    asm volatile("s_wait_asynccnt 0x0" ::: "memory");
}

// ---------------------------------------------------------------- converts
__global__ void f32_to_bf16_kernel(const float* __restrict__ in,
                                   unsigned short* __restrict__ out, int n) {
    int i = blockIdx.x * blockDim.x + threadIdx.x;
    if (i < n) out[i] = f2bf(in[i]);
}

// mel [B,T,M] f32 -> [B,M,T] bf16
__global__ void mel_transpose_kernel(const float* __restrict__ mel,
                                     unsigned short* __restrict__ out,
                                     int B, int T, int M) {
    int i = blockIdx.x * blockDim.x + threadIdx.x;
    int total = B * M * T;
    if (i >= total) return;
    int t = i % T;
    int m = (i / T) % M;
    int b = i / (T * M);
    out[i] = f2bf(mel[((size_t)b * T + t) * M + m]);
}

// ------------------------------------------------- conv1d K=5 'same' (WMMA)
// Implicit GEMM: [Cout x Cin*5] x [Cin*5 x T].  Block = 64(Cout) x 128(T),
// 8 waves in 4x2; each wave: 16x64 via 4 accumulators (A-frag reuse x4).
// A (weights) staged via async global->LDS b128; B (im2col) via reg prefetch.
// MODE 0: relu -> bf16   MODE 1: relu + residual -> bf16   MODE 2: linear -> f32
template <int MODE>
__global__ __launch_bounds__(256) void conv1d_wmma_kernel(
    const unsigned short* __restrict__ x,   // [B, Cin, T] bf16
    const unsigned short* __restrict__ w,   // [Cout, Cin*5] bf16
    const float* __restrict__ bias,         // [Cout]
    unsigned short* __restrict__ out_bf,    // MODE 0/1
    float* __restrict__ out_f,              // MODE 2
    int Cin, int Cout, int T) {
    // rows padded to 40 ushorts (80B = 20 banks): conflict-free, 16B-aligned
    __shared__ unsigned short lsA[2][64][40];   // [buf][cout 64][k 32]
    __shared__ unsigned short lsBt[2][128][40]; // [buf][t 128][k 32] transposed

    const int b    = blockIdx.z;
    const int c0   = blockIdx.y * 64;
    const int t0   = blockIdx.x * 128;
    const int tid  = threadIdx.x;
    const int wave = tid >> 5;
    const int lane = tid & 31;
    const int wr   = wave >> 1;        // 0..3 -> row tile (16 each)
    const int wc   = wave & 1;         // 0..1 -> col half (64 each)
    const int ln   = lane & 15;
    const int hi   = lane >> 4;

    const int KV = Cin * 5;
    const int KT = (KV + 31) >> 5;
    const size_t xbase = (size_t)b * Cin * T;

    union AccU { v8f v; float f[8]; } acc[4];
#pragma unroll
    for (int a = 0; a < 4; ++a)
#pragma unroll
        for (int r = 0; r < 8; ++r) acc[a].f[r] = 0.0f;

    // --- A tile async stage: 64 rows x 4 x b128 chunks = 256 transfers ---
    const int ar  = tid >> 2;
    const int ac8 = (tid & 3) << 3;
    auto loadAasync = [&](int kt, int buf) {
        unsigned lds = (unsigned)(unsigned long long)(void*)&lsA[buf][ar][ac8];
        unsigned long long g = (unsigned long long)(const void*)
            (w + (size_t)(c0 + ar) * KV + (kt << 5) + ac8);
        async_load_b128(lds, g);
    };
    // --- B tile reg prefetch: thread owns (ct, 16 consecutive k) ---
    const int ct  = tid >> 1;
    const int ck0 = (tid & 1) << 4;
    unsigned short tb[16];
    auto loadB = [&](int kt) {
        const int kk = kt << 5;
#pragma unroll
        for (int e = 0; e < 16; ++e) {
            int kq = kk + ck0 + e;
            unsigned short v = 0;
            if (kq < KV) {
                int ci = kq / 5, k5 = kq - ci * 5;
                int tg = t0 + ct + k5 - 2;
                if (tg >= 0 && tg < T) v = x[xbase + (size_t)ci * T + tg];
            }
            tb[e] = v;
        }
    };
    auto storeB = [&](int buf) {
#pragma unroll
        for (int e = 0; e < 16; ++e) lsBt[buf][ct][ck0 + e] = tb[e];
    };

    // ------------------------------- pipeline -------------------------------
    loadAasync(0, 0);
    loadB(0);
    storeB(0);
    wait_asynccnt0();
    __syncthreads();

    for (int kt = 0; kt < KT; ++kt) {
        const int buf = kt & 1, nb = buf ^ 1;
        const bool more = (kt + 1) < KT;
        if (more) { loadAasync(kt + 1, nb); loadB(kt + 1); }

        union U16 { v16u u; unsigned short s[16]; };
        U16 af;
#pragma unroll
        for (int e = 0; e < 16; ++e)   // A 16x32 bf16 layout (contig 32B run)
            af.s[e] = lsA[buf][wr * 16 + ln][hi * 16 + e];
#pragma unroll
        for (int a = 0; a < 4; ++a) {
            U16 bg;
#pragma unroll
            for (int e = 0; e < 16; ++e)  // B 32x16 layout (contig 32B run)
                bg.s[e] = lsBt[buf][wc * 64 + a * 16 + ln][hi * 16 + e];
            acc[a].v = __builtin_amdgcn_wmma_f32_16x16x32_bf16(
                false, __builtin_bit_cast(v16bf, af.u),
                false, __builtin_bit_cast(v16bf, bg.u),
                (short)0, acc[a].v, false, false);
        }
        if (more) storeB(nb);
        wait_asynccnt0();
        __syncthreads();
    }

    // epilogue per C/D layout: VGPR r -> row r + 8*hi, col = ln
#pragma unroll
    for (int a = 0; a < 4; ++a) {
#pragma unroll
        for (int r = 0; r < 8; ++r) {
            int row = c0 + wr * 16 + hi * 8 + r;
            int col = t0 + wc * 64 + a * 16 + ln;
            float v = acc[a].f[r] + bias[row];
            if constexpr (MODE != 2) { if (v < 0.0f) v = 0.0f; }
            size_t oidx = ((size_t)b * Cout + row) * T + col;
            if constexpr (MODE == 1) v += bf2f(x[oidx]);  // residual
            if constexpr (MODE == 2) out_f[oidx] = v;
            else                     out_bf[oidx] = f2bf(v);
        }
    }
}

// ------------------------------------------------------------- maxpool(8)
__global__ void maxpool8_kernel(const unsigned short* __restrict__ in,
                                unsigned short* __restrict__ out,
                                int B, int C, int T) {
    int To = T >> 3;
    int i = blockIdx.x * blockDim.x + threadIdx.x;
    int total = B * C * To;
    if (i >= total) return;
    int t = i % To;
    int c = (i / To) % C;
    int b = i / (To * C);
    size_t base = ((size_t)b * C + c) * T + (size_t)t * 8;
    float m = bf2f(in[base]);
#pragma unroll
    for (int k = 1; k < 8; ++k) {
        float v = bf2f(in[base + k]);
        m = v > m ? v : m;
    }
    out[((size_t)b * C + c) * To + t] = f2bf(m);
}

// --------------------------------------------------------------- VQ pieces
__global__ void cnorm_kernel(const float* __restrict__ cb,
                             float* __restrict__ cn) {
    int c = blockIdx.x * blockDim.x + threadIdx.x;
    if (c >= 1024) return;
    float s = 0.0f;
    for (int d = 0; d < 256; ++d) {
        float v = cb[(size_t)c * 256 + d];
        s += v * v;
    }
    cn[c] = s;
}

// scores[row, code] = z_row . codebook_code   via fp32 WMMA 16x16x4
__global__ __launch_bounds__(256) void vq_scores_kernel(
    const float* __restrict__ ze,      // [32, 256, 256]  (B, D, T')
    const float* __restrict__ cb,      // [1024, 256]
    float* __restrict__ scores) {      // [8192, 1024]
    __shared__ float lz[16][260];
    const int row0 = blockIdx.y * 16;
    const int cb0  = blockIdx.x * 128;
    const int tid  = threadIdx.x;
    const int wave = tid >> 5, lane = tid & 31;
    const int ln = lane & 15, hi = lane >> 4;
    const int n0 = cb0 + wave * 16;

    for (int i = tid; i < 16 * 256; i += 256) {
        int r = i >> 8, d = i & 255;
        int row = row0 + r;
        int b = row >> 8, t = row & 255;
        lz[r][d] = ze[((size_t)b * 256 + d) * 256 + t];
    }
    __syncthreads();

    union { v8f v; float f[8]; } acc;
#pragma unroll
    for (int r = 0; r < 8; ++r) acc.f[r] = 0.0f;

    for (int kk = 0; kk < 256; kk += 4) {
        v2f a, bg;
        a.x = lz[ln][kk + 2 * hi];
        a.y = lz[ln][kk + 2 * hi + 1];
        const float* cp = cb + (size_t)(n0 + ln) * 256 + kk + 2 * hi;
        bg.x = cp[0];
        bg.y = cp[1];
        acc.v = __builtin_amdgcn_wmma_f32_16x16x4_f32(
            false, a, false, bg, (short)0, acc.v, false, false);
    }
#pragma unroll
    for (int r = 0; r < 8; ++r) {
        int row = row0 + hi * 8 + r;
        scores[(size_t)row * 1024 + n0 + ln] = acc.f[r];
    }
}

__global__ void zero_kernel(float* p, int n) {
    int i = blockIdx.x * blockDim.x + threadIdx.x;
    if (i < n) p[i] = 0.0f;
}

// argmin over codes, gather codebook row -> zq, accumulate sum((z-q)^2)
__global__ __launch_bounds__(256) void vq_select_kernel(
    const float* __restrict__ ze, const float* __restrict__ scores,
    const float* __restrict__ cn, const float* __restrict__ cb,
    float* __restrict__ zq, float* __restrict__ loss_acc) {
    __shared__ float sd[256];
    __shared__ int   si[256];
    const int row = blockIdx.x;
    const int tid = threadIdx.x;

    float best = 3.4e38f;
    int   bi   = 0;
    for (int c = tid; c < 1024; c += 256) {
        float d = cn[c] - 2.0f * scores[(size_t)row * 1024 + c];
        if (d < best) { best = d; bi = c; }
    }
    sd[tid] = best;
    si[tid] = bi;
    __syncthreads();
    for (int s = 128; s > 0; s >>= 1) {
        if (tid < s) {
            float d2 = sd[tid + s]; int i2 = si[tid + s];
            if (d2 < sd[tid] || (d2 == sd[tid] && i2 < si[tid])) {
                sd[tid] = d2; si[tid] = i2;
            }
        }
        __syncthreads();
    }
    const int idx = si[0];
    __syncthreads();

    const int b = row >> 8, t = row & 255;
    float q  = cb[(size_t)idx * 256 + tid];
    float zv = ze[((size_t)b * 256 + tid) * 256 + t];
    zq[(size_t)row * 256 + tid] = q;
    float diff = zv - q;
    sd[tid] = diff * diff;
    __syncthreads();
    for (int s = 128; s > 0; s >>= 1) {
        if (tid < s) sd[tid] += sd[tid + s];
        __syncthreads();
    }
    if (tid == 0) atomicAdd(loss_acc, sd[0]);
}

__global__ void finalize_loss_kernel(const float* __restrict__ acc,
                                     float* __restrict__ out_tail) {
    float m = *acc * (1.0f / (8192.0f * 256.0f));
    out_tail[0] = m;  // commit_loss
    out_tail[1] = m;  // vq_loss (== commit_loss since zq == q)
}

// ------------------------------------------------------------------ driver
extern "C" void kernel_launch(void* const* d_in, const int* in_sizes, int n_in,
                              void* d_out, int out_size, void* d_ws,
                              size_t ws_size, hipStream_t stream) {
    (void)in_sizes; (void)n_in; (void)out_size; (void)ws_size;
    const float* mel    = (const float*)d_in[0];
    const float* w_pre  = (const float*)d_in[1];
    const float* b_pre  = (const float*)d_in[2];
    const float* w_b1   = (const float*)d_in[3];
    const float* b_b1   = (const float*)d_in[4];
    const float* w_b2   = (const float*)d_in[5];
    const float* b_b2   = (const float*)d_in[6];
    const float* w_post = (const float*)d_in[7];
    const float* b_post = (const float*)d_in[8];
    const float* cb     = (const float*)d_in[9];

    constexpr int B = 32, T = 2048, M = 80, H = 384, D = 256, BINS = 1024,
                  NB = 6, Tp = T / 8;

    char*  ws  = (char*)d_ws;
    size_t off = 0;
    auto alloc = [&](size_t bytes) -> void* {
        void* p = ws + off;
        off = (off + bytes + 255) & ~(size_t)255;
        return p;
    };
    unsigned short* melb = (unsigned short*)alloc((size_t)B * M * T * 2);
    unsigned short* wpre = (unsigned short*)alloc((size_t)H * M * 5 * 2);
    unsigned short* wb1  = (unsigned short*)alloc((size_t)NB * H * H * 5 * 2);
    unsigned short* wb2  = (unsigned short*)alloc((size_t)NB * H * H * 5 * 2);
    unsigned short* wpo  = (unsigned short*)alloc((size_t)D * H * 5 * 2);
    unsigned short* XA   = (unsigned short*)alloc((size_t)B * H * T * 2);
    unsigned short* XB   = (unsigned short*)alloc((size_t)B * H * T * 2);
    unsigned short* PA   = (unsigned short*)alloc((size_t)B * H * Tp * 2);
    unsigned short* PB   = (unsigned short*)alloc((size_t)B * H * Tp * 2);
    float* ze     = (float*)alloc((size_t)B * D * Tp * 4);
    float* scores = (float*)alloc((size_t)B * Tp * BINS * 4);
    float* cn     = (float*)alloc((size_t)BINS * 4);
    float* lacc   = (float*)alloc(256);

    auto cvt = [&](const float* in, unsigned short* o, int n) {
        f32_to_bf16_kernel<<<(n + 255) / 256, 256, 0, stream>>>(in, o, n);
    };
    cvt(w_pre, wpre, H * M * 5);
    cvt(w_b1, wb1, NB * H * H * 5);
    cvt(w_b2, wb2, NB * H * H * 5);
    cvt(w_post, wpo, D * H * 5);
    mel_transpose_kernel<<<(B * M * T + 255) / 256, 256, 0, stream>>>(
        mel, melb, B, T, M);

    // pre-conv: 80 -> 384, relu
    conv1d_wmma_kernel<0><<<dim3(T / 128, H / 64, B), 256, 0, stream>>>(
        melb, wpre, b_pre, XA, nullptr, M, H, T);

    unsigned short *cur = XA, *nxt = XB;
    for (int i = 0; i < NB; ++i) {
        conv1d_wmma_kernel<1><<<dim3(T / 128, H / 64, B), 256, 0, stream>>>(
            cur, wb1 + (size_t)i * H * H * 5, b_b1 + i * H, nxt, nullptr,
            H, H, T);
        unsigned short* t2 = cur; cur = nxt; nxt = t2;
    }
    maxpool8_kernel<<<(B * H * Tp + 255) / 256, 256, 0, stream>>>(
        cur, PA, B, H, T);
    unsigned short *pc = PA, *pn = PB;
    for (int i = 0; i < NB; ++i) {
        conv1d_wmma_kernel<1><<<dim3(Tp / 128, H / 64, B), 256, 0, stream>>>(
            pc, wb2 + (size_t)i * H * H * 5, b_b2 + i * H, pn, nullptr,
            H, H, Tp);
        unsigned short* t2 = pc; pc = pn; pn = t2;
    }
    // post-conv: 384 -> 256, linear, fp32 out
    conv1d_wmma_kernel<2><<<dim3(Tp / 128, D / 64, B), 256, 0, stream>>>(
        pc, wpo, b_post, nullptr, ze, H, D, Tp);

    // VQ
    cnorm_kernel<<<4, 256, 0, stream>>>(cb, cn);
    vq_scores_kernel<<<dim3(BINS / 128, (B * Tp) / 16), 256, 0, stream>>>(
        ze, cb, scores);
    zero_kernel<<<1, 64, 0, stream>>>(lacc, 1);
    float* zq = (float*)d_out;
    vq_select_kernel<<<B * Tp, 256, 0, stream>>>(ze, scores, cn, cb, zq, lacc);
    finalize_loss_kernel<<<1, 1, 0, stream>>>(lacc, zq + (size_t)B * Tp * D);
}
